// GPT2Attention_7516192768317
// MI455X (gfx1250) — compile-verified, matrix operand order checked by reading, not run on hardware
//
#include <hip/hip_runtime.h>

typedef __attribute__((ext_vector_type(16))) __bf16 v16bf;
typedef __attribute__((ext_vector_type(8)))  __bf16 v8bf;
typedef __attribute__((ext_vector_type(4)))  __bf16 v4bf;
typedef __attribute__((ext_vector_type(8)))  float  v8f;

#define HIDDEN   1024
#define HEADS    16
#define HEAD_DIM 64
#define BATCH    2
#define SEQ      2048
#define M_TOTAL  (BATCH * SEQ)   // 4096
#define KDIM     1024            // both GEMMs reduce over 1024

#define BM 128
#define BN 128
#define BK 32
#define LDSP 40                  // padded row stride (bf16 elems) to dodge bank conflicts

static __device__ __forceinline__ v16bf concat8(v8bf lo, v8bf hi) {
    return __builtin_shufflevector(lo, hi, 0,1,2,3,4,5,6,7,8,9,10,11,12,13,14,15);
}

// LDS byte offset of a __shared__ object (flat addr low 32 bits == LDS offset)
static __device__ __forceinline__ uint32_t lds_off(const void* p) {
    return (uint32_t)(uintptr_t)p;
}

// CDNA5 async copy: 16 bytes per lane, global -> LDS, tracked by ASYNCcnt.
static __device__ __forceinline__ void async_copy16(uint32_t lds_addr, const __bf16* gptr) {
    asm volatile("global_load_async_to_lds_b128 %0, %1, off"
                 :: "v"(lds_addr), "v"(gptr)
                 : "memory");
}
static __device__ __forceinline__ void wait_async0() {
    asm volatile("s_wait_asynccnt 0x0" ::: "memory");
}
static __device__ __forceinline__ void wait_ds0() {
    asm volatile("s_wait_dscnt 0x0" ::: "memory");
}

// ---------------------------------------------------------------------------
// f32 -> bf16 conversion (vectorized x4)
// ---------------------------------------------------------------------------
__global__ __launch_bounds__(256)
void cvt_f32_bf16(const float* __restrict__ in, __bf16* __restrict__ out, int n4) {
    int i = blockIdx.x * 256 + threadIdx.x;
    if (i < n4) {
        float4 v = ((const float4*)in)[i];
        v4bf o;
        o[0] = (__bf16)v.x; o[1] = (__bf16)v.y;
        o[2] = (__bf16)v.z; o[3] = (__bf16)v.w;
        ((v4bf*)out)[i] = o;
    }
}

// ---------------------------------------------------------------------------
// Tiled bf16 WMMA GEMM with double-buffered async global->LDS pipeline.
//   C[m,n] = sum_k A[m,k] * B[n,k] + bias[n]
//   mode 0: write f32 to outF[m*N + n]
//   mode 1: N==3072 qkv split -> Q,K as [B,H,S,D] bf16, V transposed [B,H,D,S]
// Block: 256 thr (8 waves), tile 128x128x32; wave = 64x32 (4x2 wmma tiles)
// ---------------------------------------------------------------------------
__global__ __launch_bounds__(256)
void gemm_bf16(const __bf16* __restrict__ A, const __bf16* __restrict__ Bw,
               const float* __restrict__ bias,
               float* __restrict__ outF,
               __bf16* __restrict__ Qo, __bf16* __restrict__ Ko, __bf16* __restrict__ Vto,
               int M, int N, int mode)
{
    __shared__ __attribute__((aligned(16))) __bf16 lA[2][BM * LDSP];
    __shared__ __attribute__((aligned(16))) __bf16 lB[2][BN * LDSP];

    const int tid  = threadIdx.x;
    const int lane = tid & 31;
    const int wave = tid >> 5;
    const int r    = lane & 15;     // row / col within 16
    const int g    = lane >> 4;     // half-wave group
    const int wm   = wave >> 2;     // 0..1
    const int wn   = wave & 3;      // 0..3
    const int m0   = blockIdx.x * BM;
    const int n0   = blockIdx.y * BN;

    // ---- per-thread async-copy mapping: 4 x 16B chunks per tile ----
    const int row0 = tid >> 2;            // rows 0..63
    const int row1 = row0 + 64;           // rows 64..127
    const int cc   = (tid & 3) * 8;       // 0,8,16,24 (bf16 elems)
    const __bf16* aP0 = A  + (size_t)(m0 + row0) * KDIM + cc;
    const __bf16* aP1 = A  + (size_t)(m0 + row1) * KDIM + cc;
    const __bf16* bP0 = Bw + (size_t)(n0 + row0) * KDIM + cc;
    const __bf16* bP1 = Bw + (size_t)(n0 + row1) * KDIM + cc;
    const uint32_t o0 = (uint32_t)(row0 * LDSP + cc) * 2u;
    const uint32_t o1 = (uint32_t)(row1 * LDSP + cc) * 2u;
    uint32_t baseA[2] = { lds_off(&lA[0][0]), lds_off(&lA[1][0]) };
    uint32_t baseB[2] = { lds_off(&lB[0][0]), lds_off(&lB[1][0]) };

    v8f acc[4][2];
    v8f zero = {};
    #pragma unroll
    for (int i = 0; i < 4; ++i)
        #pragma unroll
        for (int j = 0; j < 2; ++j) acc[i][j] = zero;

    // ---- prologue: stream tile 0 into buffer 0 ----
    async_copy16(baseA[0] + o0, aP0);  aP0 += BK;
    async_copy16(baseA[0] + o1, aP1);  aP1 += BK;
    async_copy16(baseB[0] + o0, bP0);  bP0 += BK;
    async_copy16(baseB[0] + o1, bP1);  bP1 += BK;

    const int kTiles = KDIM / BK;      // 32
    #pragma unroll 2
    for (int kt = 0; kt < kTiles; ++kt) {
        const int cur = kt & 1;
        wait_async0();                 // my chunks of tile kt are in LDS
        wait_ds0();                    // my reads of buf[cur] from iter kt-2 done
        __syncthreads();               // everyone's tile kt resident, old reads done

        // stream tile kt+1 while computing tile kt
        if (kt + 1 < kTiles) {
            const int nxt = cur ^ 1;
            async_copy16(baseA[nxt] + o0, aP0);  aP0 += BK;
            async_copy16(baseA[nxt] + o1, aP1);  aP1 += BK;
            async_copy16(baseB[nxt] + o0, bP0);  bP0 += BK;
            async_copy16(baseB[nxt] + o1, bP1);  bP1 += BK;
        }

        const __bf16* cA = &lA[cur][0];
        const __bf16* cB = &lB[cur][0];

        // B fragments (A-layout of B^T: lane r = column n, K chunks by g)
        v16bf bfr[2];
        #pragma unroll
        for (int j = 0; j < 2; ++j) {
            const __bf16* p = cB + (wn * 32 + j * 16 + r) * LDSP + g * 8;
            bfr[j] = concat8(*(const v8bf*)p, *(const v8bf*)(p + 16));
        }
        v16bf afr[4];
        #pragma unroll
        for (int i = 0; i < 4; ++i) {
            const __bf16* p = cA + (wm * 64 + i * 16 + r) * LDSP + g * 8;
            afr[i] = concat8(*(const v8bf*)p, *(const v8bf*)(p + 16));
        }
        #pragma unroll
        for (int i = 0; i < 4; ++i)
            #pragma unroll
            for (int j = 0; j < 2; ++j)
                acc[i][j] = __builtin_amdgcn_wmma_f32_16x16x32_bf16(
                    false, afr[i], false, bfr[j], (short)0, acc[i][j], false, false);
    }

    // Epilogue. C-tile layout: VGPR v, lane -> M = v + 8*g, N = r
    #pragma unroll
    for (int i = 0; i < 4; ++i) {
        #pragma unroll
        for (int j = 0; j < 2; ++j) {
            const int gn = n0 + wn * 32 + j * 16 + r;
            const float bv = bias[gn];
            #pragma unroll
            for (int v = 0; v < 8; ++v) {
                const int gm = m0 + wm * 64 + i * 16 + v + g * 8;
                float val = acc[i][j][v] + bv;
                if (mode == 0) {
                    outF[(size_t)gm * N + gn] = val;
                } else {
                    __bf16 bb = (__bf16)val;
                    int sec = gn >> 10, c = gn & 1023;
                    int h = c >> 6, d = c & 63;
                    int b = gm >> 11, s = gm & 2047;
                    if (sec == 0)
                        Qo[((size_t)(b * HEADS + h) * SEQ + s) * HEAD_DIM + d] = bb;
                    else if (sec == 1)
                        Ko[((size_t)(b * HEADS + h) * SEQ + s) * HEAD_DIM + d] = bb;
                    else
                        Vto[((size_t)(b * HEADS + h) * HEAD_DIM + d) * SEQ + s] = bb;
                }
            }
        }
    }
}

// ---------------------------------------------------------------------------
// Flash attention (causal), one wave = 16 queries of one (b,h).
// Q,K: [B,H,S,D] bf16.  Vt: [B,H,D,S] bf16.  out: [B*S, HIDDEN] bf16.
// ---------------------------------------------------------------------------
__global__ __launch_bounds__(128)
void flash_attn(const __bf16* __restrict__ Q, const __bf16* __restrict__ Kk,
                const __bf16* __restrict__ Vt, __bf16* __restrict__ out)
{
    __shared__ __attribute__((aligned(16))) __bf16 pbuf[4][16 * LDSP];

    const int tid   = threadIdx.x;
    const int lane  = tid & 31;
    const int wslot = tid >> 5;
    const int r     = lane & 15;
    const int g     = lane >> 4;
    const int w     = blockIdx.x * 4 + wslot;   // 0..4095
    const int qt    = w & 127;                  // query tile (S/16)
    const int bh    = w >> 7;                   // 0..31
    const int q0    = qt * 16;

    const __bf16* Qb = Q  + ((size_t)bh * SEQ + q0) * HEAD_DIM;
    const __bf16* Kb = Kk + (size_t)bh * SEQ * HEAD_DIM;
    const __bf16* Vb = Vt + (size_t)bh * HEAD_DIM * SEQ;
    __bf16* pb = &pbuf[wslot][0];

    // Q fragments: 16 x 64 = two 16x32 A-fragments
    v16bf qf[2];
    #pragma unroll
    for (int t = 0; t < 2; ++t) {
        const __bf16* p = Qb + r * HEAD_DIM + t * 32 + g * 8;
        qf[t] = concat8(*(const v8bf*)p, *(const v8bf*)(p + 16));
    }

    v8f o[4];
    v8f zero = {};
    #pragma unroll
    for (int dt = 0; dt < 4; ++dt) o[dt] = zero;
    float mr[8], lr[8];
    #pragma unroll
    for (int v = 0; v < 8; ++v) { mr[v] = -3.0e38f; lr[v] = 0.0f; }

    const float scale = 0.125f;                      // 1/sqrt(64)
    const float L2E   = 1.44269504088896340736f;

    const int nkb = (q0 + 16 + 31) >> 5;             // causal: keys <= q0+15
    for (int kb = 0; kb < nkb; ++kb) {
        const int kk = kb * 32;

        // S = Q @ K^T : two 16x16 tiles (keys kk..kk+15, kk+16..kk+31)
        v8f st[2];
        #pragma unroll
        for (int tile = 0; tile < 2; ++tile) {
            const __bf16* p0 = Kb + (size_t)(kk + tile * 16 + r) * HEAD_DIM + g * 8;
            v16bf kf0 = concat8(*(const v8bf*)p0, *(const v8bf*)(p0 + 16));
            const __bf16* p1 = p0 + 32;
            v16bf kf1 = concat8(*(const v8bf*)p1, *(const v8bf*)(p1 + 16));
            v8f z = zero;
            z = __builtin_amdgcn_wmma_f32_16x16x32_bf16(false, qf[0], false, kf0, (short)0, z, false, false);
            z = __builtin_amdgcn_wmma_f32_16x16x32_bf16(false, qf[1], false, kf1, (short)0, z, false, false);
            st[tile] = z;
        }

        // Online softmax: rows live in VGPR index v (+8g); columns in lanes (r)
        #pragma unroll
        for (int v = 0; v < 8; ++v) {
            const int qg = q0 + v + 8 * g;
            float x0 = st[0][v] * scale; if (kk + r > qg)      x0 = -3.0e38f;
            float x1 = st[1][v] * scale; if (kk + 16 + r > qg) x1 = -3.0e38f;
            float t = fmaxf(x0, x1);
            t = fmaxf(t, __shfl_xor(t, 1));
            t = fmaxf(t, __shfl_xor(t, 2));
            t = fmaxf(t, __shfl_xor(t, 4));
            t = fmaxf(t, __shfl_xor(t, 8));
            float mnew  = fmaxf(mr[v], t);
            float alpha = __builtin_exp2f((mr[v] - mnew) * L2E);
            float p0 = __builtin_exp2f((x0 - mnew) * L2E);
            float p1 = __builtin_exp2f((x1 - mnew) * L2E);
            float sp = p0 + p1;
            sp += __shfl_xor(sp, 1);
            sp += __shfl_xor(sp, 2);
            sp += __shfl_xor(sp, 4);
            sp += __shfl_xor(sp, 8);
            lr[v] = lr[v] * alpha + sp;
            mr[v] = mnew;
            #pragma unroll
            for (int dt = 0; dt < 4; ++dt) o[dt][v] *= alpha;
            // C-layout -> LDS row-major P tile (16 x 32)
            pb[(v + 8 * g) * LDSP + r]      = (__bf16)p0;
            pb[(v + 8 * g) * LDSP + 16 + r] = (__bf16)p1;
        }
        wait_ds0();

        // P as A-fragment, V^T columns as B-fragments
        const __bf16* pp = pb + r * LDSP + g * 8;
        v16bf pf = concat8(*(const v8bf*)pp, *(const v8bf*)(pp + 16));
        #pragma unroll
        for (int dt = 0; dt < 4; ++dt) {
            const __bf16* vp = Vb + (size_t)(dt * 16 + r) * SEQ + kk + g * 8;
            v16bf vf = concat8(*(const v8bf*)vp, *(const v8bf*)(vp + 16));
            o[dt] = __builtin_amdgcn_wmma_f32_16x16x32_bf16(
                false, pf, false, vf, (short)0, o[dt], false, false);
        }
    }

    // Normalize + write [B*S, HIDDEN] bf16
    const int b = bh >> 4, h = bh & 15;
    #pragma unroll
    for (int v = 0; v < 8; ++v) {
        float inv = 1.0f / lr[v];
        const int gm = b * SEQ + q0 + v + 8 * g;
        #pragma unroll
        for (int dt = 0; dt < 4; ++dt)
            out[(size_t)gm * HIDDEN + h * HEAD_DIM + dt * 16 + r] = (__bf16)(o[dt][v] * inv);
    }
}

// ---------------------------------------------------------------------------
// Host launcher
// ---------------------------------------------------------------------------
extern "C" void kernel_launch(void* const* d_in, const int* in_sizes, int n_in,
                              void* d_out, int out_size, void* d_ws, size_t ws_size,
                              hipStream_t stream)
{
    const float* x      = (const float*)d_in[0];
    // d_in[1] = attention_mask (causal; applied analytically in flash_attn)
    const float* W_attn = (const float*)d_in[2];
    const float* b_attn = (const float*)d_in[3];
    const float* W_proj = (const float*)d_in[4];
    const float* b_proj = (const float*)d_in[5];
    float* out = (float*)d_out;

    char* ws = (char*)d_ws;
    __bf16* xb   = (__bf16*)(ws);                 //  8 MB  [4096,1024]
    __bf16* Wab  = (__bf16*)(ws + 8388608);       //  6 MB  [3072,1024]
    __bf16* Wpb  = (__bf16*)(ws + 14680064);      //  2 MB  [1024,1024]
    __bf16* Qb   = (__bf16*)(ws + 16777216);      //  8 MB  [B,H,S,D]
    __bf16* Kb   = (__bf16*)(ws + 25165824);      //  8 MB  [B,H,S,D]
    __bf16* Vtb  = (__bf16*)(ws + 33554432);      //  8 MB  [B,H,D,S]
    __bf16* atob = (__bf16*)(ws + 41943040);      //  8 MB  [4096,1024]

    cvt_f32_bf16<<<4096, 256, 0, stream>>>(x,      xb,  4194304 / 4);
    cvt_f32_bf16<<<3072, 256, 0, stream>>>(W_attn, Wab, 3145728 / 4);
    cvt_f32_bf16<<<1024, 256, 0, stream>>>(W_proj, Wpb, 1048576 / 4);

    // QKV projection + split/transpose into Q, K, V^T
    gemm_bf16<<<dim3(32, 24), 256, 0, stream>>>(xb, Wab, b_attn,
                                                nullptr, Qb, Kb, Vtb,
                                                M_TOTAL, 3 * HIDDEN, 1);
    // Causal flash attention
    flash_attn<<<1024, 128, 0, stream>>>(Qb, Kb, Vtb, atob);

    // Output projection (f32 result)
    gemm_bf16<<<dim3(32, 8), 256, 0, stream>>>(atob, Wpb, b_proj,
                                               out, nullptr, nullptr, nullptr,
                                               M_TOTAL, HIDDEN, 0);
}
